// SAINTTransformer_72361609003514
// MI455X (gfx1250) — compile-verified, matrix operand order checked by reading, not run on hardware
//
#include <hip/hip_runtime.h>
#include <hip/hip_bf16.h>
#include <cstdint>

typedef __attribute__((ext_vector_type(8)))  _Float16 v8h;
typedef __attribute__((ext_vector_type(16))) _Float16 v16h;
typedef __attribute__((ext_vector_type(8)))  float    v8f;

#define B_   64
#define H_   15
#define D_   512
#define NH_  8
#define HD_  64
#define DFF_ 2048
#define SEQ_ 16
#define F_   32
#define T_   (B_*SEQ_*F_)   /* 32768 tokens */
#define EPS_ 1e-7f
#define BETA_ 0.05f

#define USE_ASYNC 1          /* CDNA5 global_load_async_to_lds_b128 path */

// ---------------- helpers ----------------
__device__ __forceinline__ float wredsum(float v){
#pragma unroll
  for (int o = 16; o; o >>= 1) v += __shfl_xor(v, o, 32);
  return v;
}
__device__ __forceinline__ v8f zero8(){
  v8f z;
#pragma unroll
  for (int i = 0; i < 8; ++i) z[i] = 0.f;
  return z;
}
__device__ __forceinline__ v8h zeroh8(){
  v8h z;
#pragma unroll
  for (int i = 0; i < 8; ++i) z[i] = (_Float16)0;
  return z;
}
__device__ __forceinline__ float gelu_f(float x){
  float t = tanhf(0.7978845608f * (x + 0.044715f * x * x * x));
  return 0.5f * x * (1.f + t);
}
__device__ __forceinline__ float gelu_g(float x){
  float x2 = x * x;
  float t  = tanhf(0.7978845608f * (x + 0.044715f * x * x2));
  float dt = (1.f - t * t) * 0.7978845608f * (1.f + 3.f * 0.044715f * x2);
  return 0.5f * (1.f + t) + 0.5f * x * dt;
}

#if USE_ASYNC
// CDNA5 async global->LDS copy, 16B per enabled lane; tracked by ASYNCcnt.
// LDS destination address is the low 32 bits of the flat pointer (ISA 10.2:
// LDS aperture addressing).
__device__ __forceinline__ void async_ld128(const _Float16* g, _Float16* l){
  unsigned lds = (unsigned)(uintptr_t)l;
  unsigned long long ga = (unsigned long long)(uintptr_t)g;
  asm volatile("global_load_async_to_lds_b128 %0, %1, off"
               :: "v"(lds), "v"(ga) : "memory");
}
__device__ __forceinline__ void wait_async0(){
  asm volatile("s_wait_asynccnt 0x0" ::: "memory");
}
#endif

// Load one 16x32 f16 WMMA A/B fragment.  Source rows are [m][k] (A) or
// [n][k] (B pre-transposed); per ISA: lane = m + 16*khalf, elements
// e<8 -> k = khalf*8+e, e>=8 -> k = 16+khalf*8+(e-8): two 16B loads.
__device__ __forceinline__ v16h frag_ld(const _Float16* p0, long rstride, int lane){
  int r  = lane & 15;
  int kh = (lane >> 4) & 1;
  const _Float16* p = p0 + (long)r * rstride + kh * 8;
  v8h lo = *reinterpret_cast<const v8h*>(p);
  v8h hi = *reinterpret_cast<const v8h*>(p + 16);
  v16h f;
#pragma unroll
  for (int i = 0; i < 8; ++i){ f[i] = lo[i]; f[i + 8] = hi[i]; }
  return f;
}

// ---------------- weight convert + transpose: W[K][N] f32 -> Wt[N][K] f16 ----------------
__global__ void wt_k(const float* __restrict__ W, _Float16* __restrict__ out, int K, int N){
  long i = (long)blockIdx.x * 256 + threadIdx.x;
  if (i >= (long)K * N) return;
  int k = (int)(i / N), n = (int)(i % N);
  out[(long)n * K + k] = (_Float16)W[i];
}

// ---------------- embedding -> z fp32 [T][512] ----------------
__global__ void embed_k(const float* __restrict__ x_cont, const int* __restrict__ x_cat,
                        const float* __restrict__ emb_w, const float* __restrict__ emb_b,
                        const float* __restrict__ cat_emb, const float* __restrict__ cls,
                        float* __restrict__ z){
  long i = (long)blockIdx.x * 256 + threadIdx.x;
  if (i >= (long)T_ * D_) return;
  int d = (int)(i & (D_ - 1));
  long tok = i >> 9;
  int f  = (int)(tok & 31);
  int bs = (int)(tok >> 5);
  int s  = bs & 15, b = bs >> 4;
  float val;
  if (s == H_) {
    val = cls[(long)f * D_ + d];
  } else if (f < 24) {
    val = x_cont[((long)b * H_ + s) * 24 + f] * emb_w[(long)f * D_ + d] + emb_b[(long)f * D_ + d];
  } else {
    int c = x_cat[((long)b * H_ + s) * 8 + (f - 24)];
    val = cat_emb[(((long)(f - 24) * 50) + c) * D_ + d];
  }
  z[i] = val;
}

// ---------------- layernorm: fp32 [rows][512] -> f16 ----------------
__global__ __launch_bounds__(256) void ln_k(const float* __restrict__ x, const float* __restrict__ g,
                                            const float* __restrict__ bta, _Float16* __restrict__ out,
                                            int rows){
  int row  = blockIdx.x * 8 + (threadIdx.x >> 5);
  int lane = threadIdx.x & 31;
  if (row >= rows) return;
  const float* xr = x + (long)row * D_;
  float vals[16]; float s = 0.f;
#pragma unroll
  for (int i = 0; i < 16; ++i){ vals[i] = xr[lane + i * 32]; s += vals[i]; }
  s = wredsum(s);
  float mean = s * (1.f / 512.f);
  float vv = 0.f;
#pragma unroll
  for (int i = 0; i < 16; ++i){ float d = vals[i] - mean; vv += d * d; }
  vv = wredsum(vv);
  float rstd = rsqrtf(vv * (1.f / 512.f) + 1e-5f);
  _Float16* o = out + (long)row * D_;
#pragma unroll
  for (int i = 0; i < 16; ++i){
    int c = lane + i * 32;
    o[c] = (_Float16)((vals[i] - mean) * rstd * g[c] + bta[c]);
  }
}

// ---------------- tiled WMMA GEMM: C[M,N] = A[M,K] * Bt[N,K]^T (+bias,+act,+resid) ----------------
// 128x128 block tile, 8 waves x (32x64), double-buffered LDS, async global->LDS fill.
__global__ __launch_bounds__(256) void gemm_k(const _Float16* __restrict__ A, const _Float16* __restrict__ Bt,
                                              const float* __restrict__ bias, const float* __restrict__ resid,
                                              float* __restrict__ Cf, _Float16* __restrict__ Ch,
                                              int M, int N, int K, int act){
  __shared__ __align__(16) _Float16 sA[2][128 * 48];
  __shared__ __align__(16) _Float16 sB[2][128 * 48];
  const int bm = blockIdx.y * 128, bn = blockIdx.x * 128;
  const int tid = threadIdx.x;
  const int lane = tid & 31, wave = tid >> 5;
  const int wrow = (wave & 3) * 32, wcol = (wave >> 2) * 64;
  v8f acc[2][4];
#pragma unroll
  for (int i = 0; i < 2; ++i)
#pragma unroll
    for (int j = 0; j < 4; ++j) acc[i][j] = zero8();

  // each thread owns 2 (row, col8) slots of the 128x32 tile (512 vec8 total)
  const int v0 = tid,        r0 = v0 >> 2, c0 = (v0 & 3) * 8;
  const int v1 = tid + 256,  r1 = v1 >> 2, c1 = (v1 & 3) * 8;
  const int nk = K >> 5;
  const bool full = (bm + 128 <= M) && (bn + 128 <= N);

#if USE_ASYNC
  if (full){
    // ---- async double-buffered pipeline ----
    {
      // issue tile 0 into buffer 0
      async_ld128(A  + (long)(bm + r0) * K + c0, &sA[0][r0 * 48 + c0]);
      async_ld128(A  + (long)(bm + r1) * K + c1, &sA[0][r1 * 48 + c1]);
      async_ld128(Bt + (long)(bn + r0) * K + c0, &sB[0][r0 * 48 + c0]);
      async_ld128(Bt + (long)(bn + r1) * K + c1, &sB[0][r1 * 48 + c1]);
    }
    for (int it = 0; it < nk; ++it){
      const int cur = it & 1;
      wait_async0();
      __syncthreads();                 // buf[cur] complete; prior reads of buf[cur^1] done
      if (it + 1 < nk){
        const int k0 = (it + 1) * 32;
        async_ld128(A  + (long)(bm + r0) * K + k0 + c0, &sA[cur ^ 1][r0 * 48 + c0]);
        async_ld128(A  + (long)(bm + r1) * K + k0 + c1, &sA[cur ^ 1][r1 * 48 + c1]);
        async_ld128(Bt + (long)(bn + r0) * K + k0 + c0, &sB[cur ^ 1][r0 * 48 + c0]);
        async_ld128(Bt + (long)(bn + r1) * K + k0 + c1, &sB[cur ^ 1][r1 * 48 + c1]);
      }
      if (it + 2 < nk){
        const int k2 = (it + 2) * 32;
        __builtin_prefetch(A  + (long)(bm + r0) * K + k2 + c0, 0, 3);
        __builtin_prefetch(Bt + (long)(bn + r0) * K + k2 + c0, 0, 3);
      }
      v16h af[2], bf[4];
#pragma unroll
      for (int i = 0; i < 2; ++i) af[i] = frag_ld(&sA[cur][(wrow + i * 16) * 48], 48, lane);
#pragma unroll
      for (int j = 0; j < 4; ++j) bf[j] = frag_ld(&sB[cur][(wcol + j * 16) * 48], 48, lane);
#pragma unroll
      for (int i = 0; i < 2; ++i)
#pragma unroll
        for (int j = 0; j < 4; ++j)
          acc[i][j] = __builtin_amdgcn_wmma_f32_16x16x32_f16(false, af[i], false, bf[j],
                                                             (short)0, acc[i][j], false, false);
    }
  } else
#endif
  {
    // ---- guarded synchronous fallback (edge tiles) ----
    for (int k0 = 0; k0 < K; k0 += 32){
#pragma unroll
      for (int i = 0; i < 2; ++i){
        int v = tid + i * 256;
        int r = v >> 2, c = (v & 3) * 8;
        v8h av = zeroh8();
        int gr = bm + r;
        if (gr < M) av = *reinterpret_cast<const v8h*>(A + (long)gr * K + k0 + c);
        *reinterpret_cast<v8h*>(&sA[0][r * 48 + c]) = av;
        v8h bv = zeroh8();
        int gn = bn + r;
        if (gn < N) bv = *reinterpret_cast<const v8h*>(Bt + (long)gn * K + k0 + c);
        *reinterpret_cast<v8h*>(&sB[0][r * 48 + c]) = bv;
      }
      __syncthreads();
      v16h af[2], bf[4];
#pragma unroll
      for (int i = 0; i < 2; ++i) af[i] = frag_ld(&sA[0][(wrow + i * 16) * 48], 48, lane);
#pragma unroll
      for (int j = 0; j < 4; ++j) bf[j] = frag_ld(&sB[0][(wcol + j * 16) * 48], 48, lane);
#pragma unroll
      for (int i = 0; i < 2; ++i)
#pragma unroll
        for (int j = 0; j < 4; ++j)
          acc[i][j] = __builtin_amdgcn_wmma_f32_16x16x32_f16(false, af[i], false, bf[j],
                                                             (short)0, acc[i][j], false, false);
      __syncthreads();
    }
  }

  const int kh = lane >> 4, ln = lane & 15;
#pragma unroll
  for (int i = 0; i < 2; ++i){
#pragma unroll
    for (int j = 0; j < 4; ++j){
      int col = bn + wcol + j * 16 + ln;
      if (col >= N) continue;
      float bc = bias ? bias[col] : 0.f;
#pragma unroll
      for (int r = 0; r < 8; ++r){
        int row = bm + wrow + i * 16 + kh * 8 + r;
        if (row >= M) continue;
        float vv = acc[i][j][r] + bc;
        if (act == 1) vv = gelu_f(vv);
        else if (act == 2) vv = tanhf(vv);
        long o = (long)row * N + col;
        if (resid) vv += resid[o];
        if (Cf) Cf[o] = vv;
        if (Ch) Ch[o] = (_Float16)vv;
      }
    }
  }
}

// ---------------- feature attention: 32 tokens, per (slice, head) wave ----------------
__global__ __launch_bounds__(32) void attn_fa_k(const _Float16* __restrict__ qkv, _Float16* __restrict__ out){
  __shared__ __align__(16) float    sS[32][33];
  __shared__ __align__(16) _Float16 sP[32][48];
  __shared__ __align__(16) _Float16 sVt[64][48];
  int slice = blockIdx.x;            // b*16+s : 0..1023
  int head  = blockIdx.y;            // 0..7
  int lane  = threadIdx.x;
  long t0 = (long)slice * F_;
  const _Float16* q = qkv + t0 * 1536 + head * HD_;
  const _Float16* k = qkv + t0 * 1536 + 512 + head * HD_;
  const _Float16* v = qkv + t0 * 1536 + 1024 + head * HD_;
  const int kh = lane >> 4, ln = lane & 15;

  v8f sacc[2][2];
#pragma unroll
  for (int i = 0; i < 2; ++i)
#pragma unroll
    for (int j = 0; j < 2; ++j) sacc[i][j] = zero8();
#pragma unroll
  for (int kk = 0; kk < 2; ++kk){
    v16h af[2], bf[2];
    af[0] = frag_ld(q + kk * 32, 1536, lane);
    af[1] = frag_ld(q + (long)16 * 1536 + kk * 32, 1536, lane);
    bf[0] = frag_ld(k + kk * 32, 1536, lane);
    bf[1] = frag_ld(k + (long)16 * 1536 + kk * 32, 1536, lane);
#pragma unroll
    for (int i = 0; i < 2; ++i)
#pragma unroll
      for (int j = 0; j < 2; ++j)
        sacc[i][j] = __builtin_amdgcn_wmma_f32_16x16x32_f16(false, af[i], false, bf[j],
                                                            (short)0, sacc[i][j], false, false);
  }
#pragma unroll
  for (int i = 0; i < 2; ++i)
#pragma unroll
    for (int j = 0; j < 2; ++j)
#pragma unroll
      for (int r = 0; r < 8; ++r)
        sS[i * 16 + kh * 8 + r][j * 16 + ln] = sacc[i][j][r] * 0.125f;
  __syncthreads();

  { // softmax over 32 cols, one row per lane
    float rowv[32]; float mx = -1e30f;
#pragma unroll
    for (int c = 0; c < 32; ++c){ rowv[c] = sS[lane][c]; mx = fmaxf(mx, rowv[c]); }
    float sum = 0.f;
#pragma unroll
    for (int c = 0; c < 32; ++c){ rowv[c] = expf(rowv[c] - mx); sum += rowv[c]; }
    float inv = 1.f / sum;
#pragma unroll
    for (int c = 0; c < 32; ++c) sP[lane][c] = (_Float16)(rowv[c] * inv);
  }
  // transpose V (32 tokens x 64 dims) into sVt[d][t]
  for (int i = lane; i < 32 * 8; i += 32){
    int t = i >> 3, c = (i & 7) * 8;
    v8h vv = *reinterpret_cast<const v8h*>(v + (long)t * 1536 + c);
#pragma unroll
    for (int j = 0; j < 8; ++j) sVt[c + j][t] = vv[j];
  }
  __syncthreads();

#pragma unroll
  for (int mi = 0; mi < 2; ++mi){
    v16h pa = frag_ld(&sP[mi * 16][0], 48, lane);
#pragma unroll
    for (int nj = 0; nj < 4; ++nj){
      v16h vb = frag_ld(&sVt[nj * 16][0], 48, lane);
      v8f o = zero8();
      o = __builtin_amdgcn_wmma_f32_16x16x32_f16(false, pa, false, vb, (short)0, o, false, false);
#pragma unroll
      for (int r = 0; r < 8; ++r){
        long row = t0 + mi * 16 + kh * 8 + r;
        out[row * D_ + head * HD_ + nj * 16 + ln] = (_Float16)o[r];
      }
    }
  }
}

// ---------------- horse attention: 16 tokens (masked), per (slice, head) wave ----------------
__global__ __launch_bounds__(32) void attn_ha_k(const _Float16* __restrict__ qkv,
                                                const float* __restrict__ amask,
                                                _Float16* __restrict__ out){
  __shared__ __align__(16) float    sS[16][17];
  __shared__ __align__(16) _Float16 sP[16][48];
  __shared__ __align__(16) _Float16 sVt[64][48];
  int slice = blockIdx.x;            // b*32+f : 0..2047
  int head  = blockIdx.y;
  int lane  = threadIdx.x;
  int b = slice >> 5, f = slice & 31;
  long base = (long)b * SEQ_ * F_ + f;    // token s -> row base + s*F_
  long rstr = (long)F_ * 1536;
  const _Float16* q = qkv + base * 1536 + head * HD_;
  const _Float16* k = qkv + base * 1536 + 512 + head * HD_;
  const _Float16* v = qkv + base * 1536 + 1024 + head * HD_;
  const int kh = lane >> 4, ln = lane & 15;

  v8f sacc = zero8();
#pragma unroll
  for (int kk = 0; kk < 2; ++kk){
    v16h af = frag_ld(q + kk * 32, rstr, lane);
    v16h bf = frag_ld(k + kk * 32, rstr, lane);
    sacc = __builtin_amdgcn_wmma_f32_16x16x32_f16(false, af, false, bf, (short)0, sacc, false, false);
  }
#pragma unroll
  for (int r = 0; r < 8; ++r) sS[kh * 8 + r][ln] = sacc[r] * 0.125f;
  // zero pads: P cols 16..31, Vt tokens 16..31
  for (int i = lane; i < 256; i += 32){ sP[i >> 4][16 + (i & 15)] = (_Float16)0; }
  for (int i = lane; i < 1024; i += 32){ sVt[i >> 4][16 + (i & 15)] = (_Float16)0; }
  __syncthreads();

  if (lane < 16){
    float rowv[16]; float mx = -1e30f;
#pragma unroll
    for (int c = 0; c < 16; ++c){
      float mval = (c < H_) ? amask[b * H_ + c] : 1.f;
      float sv = sS[lane][c];
      sv = (mval > 0.f) ? sv : -1e9f;
      rowv[c] = sv; mx = fmaxf(mx, sv);
    }
    float sum = 0.f;
#pragma unroll
    for (int c = 0; c < 16; ++c){ rowv[c] = expf(rowv[c] - mx); sum += rowv[c]; }
    float inv = 1.f / sum;
#pragma unroll
    for (int c = 0; c < 16; ++c) sP[lane][c] = (_Float16)(rowv[c] * inv);
  }
  for (int i = lane; i < 16 * 8; i += 32){
    int t = i >> 3, c = (i & 7) * 8;
    v8h vv = *reinterpret_cast<const v8h*>(v + (long)t * rstr + c);
#pragma unroll
    for (int j = 0; j < 8; ++j) sVt[c + j][t] = vv[j];
  }
  __syncthreads();

  v16h pa = frag_ld(&sP[0][0], 48, lane);
#pragma unroll
  for (int nj = 0; nj < 4; ++nj){
    v16h vb = frag_ld(&sVt[nj * 16][0], 48, lane);
    v8f o = zero8();
    o = __builtin_amdgcn_wmma_f32_16x16x32_f16(false, pa, false, vb, (short)0, o, false, false);
#pragma unroll
    for (int r = 0; r < 8; ++r){
      int s = kh * 8 + r;
      out[(base + (long)s * F_) * D_ + head * HD_ + nj * 16 + ln] = (_Float16)o[r];
    }
  }
}

// ---------------- misc small kernels ----------------
__global__ void cvt16_k(const float* __restrict__ in, _Float16* __restrict__ out, long n){
  long i = (long)blockIdx.x * 256 + threadIdx.x;
  if (i < n) out[i] = (_Float16)in[i];
}

__global__ __launch_bounds__(256) void pooldot_k(const _Float16* __restrict__ th,
                                                 const float* __restrict__ pv, float* __restrict__ t){
  int row  = blockIdx.x * 8 + (threadIdx.x >> 5);
  int lane = threadIdx.x & 31;
  if (row >= T_) return;
  const _Float16* r = th + (long)row * D_;
  float s = 0.f;
#pragma unroll
  for (int i = 0; i < 16; ++i){ int c = lane + i * 32; s += (float)r[c] * pv[c]; }
  s = wredsum(s);
  if (lane == 0) t[row] = s;
}

__global__ __launch_bounds__(256) void pool_k(const float* __restrict__ z, const float* __restrict__ t,
                                              float* __restrict__ hf){
  __shared__ float w[32];
  int bs = blockIdx.x;               // 0..1023
  int tid = threadIdx.x;
  if (tid < 32){
    float sv = t[bs * 32 + tid];
    float mx = sv;
#pragma unroll
    for (int o = 16; o; o >>= 1) mx = fmaxf(mx, __shfl_xor(mx, o, 32));
    float e = expf(sv - mx);
    float sum = e;
#pragma unroll
    for (int o = 16; o; o >>= 1) sum += __shfl_xor(sum, o, 32);
    w[tid] = e / sum;
  }
  __syncthreads();
  for (int d = tid; d < D_; d += 256){
    float acc = 0.f;
#pragma unroll 8
    for (int f2 = 0; f2 < 32; ++f2) acc += w[f2] * z[((long)bs * 32 + f2) * D_ + d];
    hf[(long)bs * D_ + d] = acc;
  }
}

__global__ void comb_k(const float* __restrict__ hf, _Float16* __restrict__ out){
  long i = (long)blockIdx.x * 256 + threadIdx.x;
  if (i >= (long)960 * 1024) return;
  int c = (int)(i & 1023);
  int row = (int)(i >> 10);
  int b = row / 15, h = row % 15;
  float v = (c < 512) ? hf[((long)b * 16 + h) * 512 + c]
                      : hf[((long)b * 16 + 15) * 512 + (c - 512)];
  out[i] = (_Float16)v;
}

__global__ void featmask_k(const float* __restrict__ feat, const float* __restrict__ amask,
                           _Float16* __restrict__ out){
  long i = (long)blockIdx.x * 256 + threadIdx.x;
  if (i >= (long)960 * 512) return;
  out[i] = (_Float16)(feat[i] * amask[i >> 9]);
}

// ---------------- MCMC refinement (separable energy) ----------------
__global__ __launch_bounds__(256) void mcmc_k(const float* __restrict__ base, const float* __restrict__ ew1,
                                              const float* __restrict__ ew2, const float* __restrict__ amask,
                                              const float* __restrict__ p0, const int* __restrict__ ns,
                                              float* __restrict__ out){
  int row  = blockIdx.x * 8 + (threadIdx.x >> 5);
  int lane = threadIdx.x & 31;
  if (row >= B_ * H_) return;
  int b = row / H_;
  const float* wp = ew1 + (long)512 * 512;   // last row of e_w1 = d/dp weights
  float bs[16], wpv[16], w2v[16];
#pragma unroll
  for (int i = 0; i < 16; ++i){
    int c = lane + i * 32;
    bs[i]  = base[(long)row * 512 + c];
    wpv[i] = wp[c];
    w2v[i] = ew2[c];
  }
  float hpr = 0.f;
  for (int j = 0; j < H_; ++j) hpr += amask[b * H_ + j];
  float m = amask[row];
  float p = p0[row];
  int NS = ns[0];
  float coef = m / (hpr * (float)B_);
  for (int st = 0; st < NS; ++st){
    float s = 0.f;
#pragma unroll
    for (int i = 0; i < 16; ++i){
      float u = bs[i] + p * wpv[i];
      s += gelu_g(u) * wpv[i] * w2v[i];
    }
    s = wredsum(s);
    float dent = -( logf(p + EPS_) + p / (p + EPS_)
                  - logf(1.f - p + EPS_) - (1.f - p) / (1.f - p + EPS_) );
    float grad = coef * s - BETA_ * coef * dent;
    p = p - 0.1f * grad * m;
    p = (tanhf(p - 0.5f) + 1.f) * 0.5f * (1.f - 2e-7f) + 1e-7f;
    p = fminf(fmaxf(p, EPS_), 1.f - EPS_);
  }
  if (lane == 0) out[row] = p;
}

// ---------------- host orchestration ----------------
extern "C" void kernel_launch(void* const* d_in, const int* in_sizes, int n_in,
                              void* d_out, int out_size, void* d_ws, size_t ws_size,
                              hipStream_t stream){
  (void)in_sizes; (void)n_in; (void)out_size; (void)ws_size;
  const float* x_cont   = (const float*)d_in[0];
  const int*   x_cat    = (const int*)  d_in[1];
  const float* amask    = (const float*)d_in[2];
  const float* pred_init= (const float*)d_in[3];
  const float* emb_w    = (const float*)d_in[4];
  const float* emb_b    = (const float*)d_in[5];
  const float* cat_emb  = (const float*)d_in[6];
  const float* cls      = (const float*)d_in[7];
  const float* fa_wqkv  = (const float*)d_in[8];
  const float* fa_wo    = (const float*)d_in[9];
  const float* ha_wqkv  = (const float*)d_in[10];
  const float* ha_wo    = (const float*)d_in[11];
  const float* ffn_w1   = (const float*)d_in[12];
  const float* ffn_b1   = (const float*)d_in[13];
  const float* ffn_w2   = (const float*)d_in[14];
  const float* ffn_b2   = (const float*)d_in[15];
  const float* ln1_g    = (const float*)d_in[16];
  const float* ln1_b    = (const float*)d_in[17];
  const float* ln2_g    = (const float*)d_in[18];
  const float* ln2_b    = (const float*)d_in[19];
  const float* ln3_g    = (const float*)d_in[20];
  const float* ln3_b    = (const float*)d_in[21];
  const float* pool_w   = (const float*)d_in[22];
  const float* pool_b   = (const float*)d_in[23];
  const float* pool_v   = (const float*)d_in[24];
  const float* proj_w   = (const float*)d_in[25];
  const float* proj_b   = (const float*)d_in[26];
  const float* e_w1     = (const float*)d_in[27];
  const float* e_b1     = (const float*)d_in[28];
  const float* e_w2     = (const float*)d_in[29];
  const int*   nsteps   = (const int*)  d_in[31];
  float* outp = (float*)d_out;

  char* ws = (char*)d_ws;
  size_t off = 0;
  auto alloc = [&](size_t bytes)->char*{
    char* p = ws + off;
    off += (bytes + 255) & ~(size_t)255;
    return p;
  };
  float*     z    = (float*)    alloc((size_t)T_ * D_ * 4);      // activations fp32
  _Float16*  x16  = (_Float16*) alloc((size_t)T_ * D_ * 2);      // LN out / attn out f16
  _Float16*  h16  = (_Float16*) alloc((size_t)T_ * DFF_ * 2);    // qkv / ffn-hidden f16
  _Float16*  w16  = (_Float16*) alloc((size_t)17825792 * 2);     // transposed f16 weights
  float*     tbuf = (float*)    alloc((size_t)T_ * 4);           // pool scores
  float*     hf   = (float*)    alloc((size_t)1024 * D_ * 4);    // pooled per (b,s)
  _Float16*  c16  = (_Float16*) alloc((size_t)960 * 1024 * 2);   // combined f16
  float*     feat = (float*)    alloc((size_t)960 * D_ * 4);     // features fp32
  _Float16*  f16b = (_Float16*) alloc((size_t)960 * D_ * 2);     // masked features f16
  float*     baseb= (float*)    alloc((size_t)960 * D_ * 4);     // energy-net base

  const size_t LQ = 1536 * 512, LO = 512 * 512, LW1 = 2048 * 512, LW2 = 512 * 2048;
  const size_t LSTRIDE = LQ + LO + LQ + LO + LW1 + LW2;

  auto wt = [&](const float* W, _Float16* out, int K, int N){
    long n = (long)K * N;
    wt_k<<<dim3((unsigned)((n + 255) / 256)), 256, 0, stream>>>(W, out, K, N);
  };
  for (int l = 0; l < 4; ++l){
    _Float16* wb = w16 + (size_t)l * LSTRIDE;
    wt(fa_wqkv + (size_t)l * LQ,  wb,                          512, 1536);
    wt(fa_wo   + (size_t)l * LO,  wb + LQ,                     512, 512);
    wt(ha_wqkv + (size_t)l * LQ,  wb + LQ + LO,                512, 1536);
    wt(ha_wo   + (size_t)l * LO,  wb + LQ + LO + LQ,           512, 512);
    wt(ffn_w1  + (size_t)l * LW1, wb + LQ + LO + LQ + LO,      512, 2048);
    wt(ffn_w2  + (size_t)l * LW2, wb + LQ + LO + LQ + LO + LW1, 2048, 512);
  }
  _Float16* poolwt = w16 + 4 * LSTRIDE;
  _Float16* projwt = poolwt + LO;
  _Float16* ew1t   = projwt + (size_t)512 * 1024;
  wt(pool_w, poolwt, 512, 512);
  wt(proj_w, projwt, 1024, 512);
  wt(e_w1,   ew1t,   512, 512);   // first 512 rows only; last row handled in mcmc_k

  {
    long n = (long)T_ * D_;
    embed_k<<<dim3((unsigned)((n + 255) / 256)), 256, 0, stream>>>(x_cont, x_cat, emb_w, emb_b, cat_emb, cls, z);
  }

  auto gemm = [&](const _Float16* A, const _Float16* Bt, const float* bias, const float* resid,
                  float* Cf, _Float16* Ch, int M, int N, int K, int act){
    dim3 g((unsigned)((N + 127) / 128), (unsigned)((M + 127) / 128));
    gemm_k<<<g, 256, 0, stream>>>(A, Bt, bias, resid, Cf, Ch, M, N, K, act);
  };

  for (int l = 0; l < 4; ++l){
    _Float16* wb = w16 + (size_t)l * LSTRIDE;
    _Float16* wfaqkv = wb;
    _Float16* wfao   = wb + LQ;
    _Float16* whaqkv = wb + LQ + LO;
    _Float16* whao   = wb + LQ + LO + LQ;
    _Float16* ww1    = wb + LQ + LO + LQ + LO;
    _Float16* ww2    = ww1 + LW1;
    // feature attention
    ln_k<<<T_ / 8, 256, 0, stream>>>(z, ln1_g + l * 512, ln1_b + l * 512, x16, T_);
    gemm(x16, wfaqkv, nullptr, nullptr, nullptr, h16, T_, 1536, 512, 0);
    attn_fa_k<<<dim3(1024, 8), 32, 0, stream>>>(h16, x16);
    gemm(x16, wfao, nullptr, z, z, nullptr, T_, 512, 512, 0);
    // horse attention (masked)
    ln_k<<<T_ / 8, 256, 0, stream>>>(z, ln2_g + l * 512, ln2_b + l * 512, x16, T_);
    gemm(x16, whaqkv, nullptr, nullptr, nullptr, h16, T_, 1536, 512, 0);
    attn_ha_k<<<dim3(2048, 8), 32, 0, stream>>>(h16, amask, x16);
    gemm(x16, whao, nullptr, z, z, nullptr, T_, 512, 512, 0);
    // FFN
    ln_k<<<T_ / 8, 256, 0, stream>>>(z, ln3_g + l * 512, ln3_b + l * 512, x16, T_);
    gemm(x16, ww1, ffn_b1 + l * 2048, nullptr, nullptr, h16, T_, 2048, 512, 1);
    gemm(h16, ww2, ffn_b2 + l * 512, z, z, nullptr, T_, 512, 2048, 0);
  }

  // attention pooling
  cvt16_k<<<dim3((unsigned)(((long)T_ * D_ + 255) / 256)), 256, 0, stream>>>(z, x16, (long)T_ * D_);
  gemm(x16, poolwt, pool_b, nullptr, nullptr, h16, T_, 512, 512, 2);   // tanh(z@pool_w+b) f16
  pooldot_k<<<T_ / 8, 256, 0, stream>>>(h16, pool_v, tbuf);
  pool_k<<<1024, 256, 0, stream>>>(z, tbuf, hf);

  // projection + energy-net base
  comb_k<<<dim3((unsigned)(((long)960 * 1024 + 255) / 256)), 256, 0, stream>>>(hf, c16);
  gemm(c16, projwt, proj_b, nullptr, feat, nullptr, 960, 512, 1024, 0);
  featmask_k<<<dim3((unsigned)(((long)960 * 512 + 255) / 256)), 256, 0, stream>>>(feat, amask, f16b);
  gemm(f16b, ew1t, e_b1, nullptr, baseb, nullptr, 960, 512, 512, 0);

  // MCMC refinement -> output p (B,H,1)
  mcmc_k<<<120, 256, 0, stream>>>(baseb, e_w1, e_w2, amask, pred_init, nsteps, outp);
}
